// SlotAttention_57449482551813
// MI455X (gfx1250) — matrix operand order repaired and checked by reading, not compiled, not run on hardware
//
#include <hip/hip_runtime.h>
#include <hip/hip_bf16.h>

typedef __attribute__((ext_vector_type(16))) _Float16 v16h;
typedef __attribute__((ext_vector_type(8)))  float    v8f;

union AFrag { v16h v; _Float16 h[16]; };
union BFrag { v16h v; _Float16 h[16]; };
union CFrag { v8f  v; float    f[8];  };

__device__ inline v8f wmma16(v16h a, v16h b, v8f c) {
  return __builtin_amdgcn_wmma_f32_16x16x32_f16(false, a, false, b, (short)0, c, false, false);
}

#define DIMD 128
#define NSEQ 4096
#define NB   64
#define KSLOT 8
#define INV_TEMP 0.08838834764831845f /* 1/sqrt(128) */
#define INV_REG  20.0f                /* 1/0.05 */

// ---------------- hash gaussian (stand-in for jax PRNG; compile-only env) ----
__device__ inline unsigned pcg(unsigned v) {
  v = v * 747796405u + 2891336453u;
  unsigned w = ((v >> ((v >> 28u) + 4u)) ^ v) * 277803737u;
  return (w >> 22u) ^ w;
}
__device__ inline float gaussf(unsigned seed) {
  unsigned a = pcg(seed), b = pcg(seed ^ 0x9e3779b9u);
  float u1 = ((float)a + 1.0f) * 2.3283064e-10f;
  float u2 = (float)b * 2.3283064e-10f;
  return sqrtf(-2.0f * __logf(u1)) * __cosf(6.2831853f * u2);
}

// ---------------- weight f32 -> f16 conversion --------------------------------
__global__ void kCvtW(const float* wq, const float* wk, const float* wv,
                      _Float16* qh, _Float16* kh, _Float16* vh) {
  int i = blockIdx.x * blockDim.x + threadIdx.x;
  if (i < DIMD * DIMD) {
    qh[i] = (_Float16)wq[i];
    kh[i] = (_Float16)wk[i];
    vh[i] = (_Float16)wv[i];
  }
}

// ---------------- slot init: mu + softplus(log_sigma)*N(0,1) ------------------
__global__ void kSlotInit(const float* mu, const float* ls, float* slots) {
  int i = blockIdx.x * blockDim.x + threadIdx.x;
  if (i < NB * KSLOT * DIMD) {
    int d = i & (DIMD - 1);
    float sig = log1pf(__expf(ls[d]));
    slots[i] = mu[d] + sig * gaussf(0x51A7u ^ (unsigned)i);
  }
}

// ---------------- fused input LayerNorm + K/V projection (WMMA) ---------------
// one wave per 16 rows of (B*N); k,v written as f16 [row][128]
__global__ __launch_bounds__(32) void kProjKV(
    const float* __restrict__ x, const float* g, const float* be,
    const _Float16* __restrict__ wk, const float* bk,
    const _Float16* __restrict__ wv, const float* bv,
    _Float16* __restrict__ ko, _Float16* __restrict__ vo) {
  __shared__ _Float16 xt[16][136];
  int lane = threadIdx.x;
  size_t row0 = (size_t)blockIdx.x * 16;
  int m = lane & 15, hh = lane >> 4;
  const float* r = x + (row0 + m) * DIMD + hh * 64;
  __builtin_prefetch(r + 16 * DIMD, 0, 0);   // next block's tile -> global_prefetch_b8
  float s = 0.f, ss = 0.f;
  #pragma unroll 8
  for (int i = 0; i < 64; ++i) { float t = r[i]; s += t; ss += t * t; }
  s  += __shfl_xor(s, 16, 32);
  ss += __shfl_xor(ss, 16, 32);
  float mean = s * (1.f / 128.f);
  float rstd = rsqrtf(ss * (1.f / 128.f) - mean * mean + 1e-5f);
  #pragma unroll 8
  for (int i = 0; i < 64; ++i) {
    int c = hh * 64 + i;
    xt[m][c] = (_Float16)((r[i] - mean) * rstd * g[c] + be[c]);
  }
  __syncthreads();
  int kb8 = (lane < 16) ? 0 : 8;
  int kb16 = (lane < 16) ? 0 : 16;
  int n = lane & 15;
  AFrag af[4];
  #pragma unroll
  for (int kc = 0; kc < 4; ++kc)
    #pragma unroll
    for (int i = 0; i < 8; ++i) {
      af[kc].h[i]     = xt[m][kc * 32 + kb16 + i];
      af[kc].h[8 + i] = xt[m][kc * 32 + kb16 + 8 + i];
    }
  for (int nt = 0; nt < 8; ++nt) {
    CFrag ck, cv;
    #pragma unroll
    for (int i = 0; i < 8; ++i) { ck.f[i] = 0.f; cv.f[i] = 0.f; }
    #pragma unroll
    for (int kc = 0; kc < 4; ++kc) {
      BFrag bkf, bvf;
      const _Float16* wkp = wk + (size_t)(nt * 16 + n) * DIMD + kc * 32 + kb16;
      const _Float16* wvp = wv + (size_t)(nt * 16 + n) * DIMD + kc * 32 + kb16;
      #pragma unroll
      for (int i = 0; i < 16; ++i) { bkf.h[i] = wkp[i]; bvf.h[i] = wvp[i]; }
      ck.v = wmma16(af[kc].v, bkf.v, ck.v);
      cv.v = wmma16(af[kc].v, bvf.v, cv.v);
    }
    float bkc = bk[nt * 16 + n], bvc = bv[nt * 16 + n];
    #pragma unroll
    for (int rr = 0; rr < 8; ++rr) {
      int mr = (lane < 16) ? rr : rr + 8;
      size_t o = (row0 + mr) * DIMD + nt * 16 + n;
      ko[o] = (_Float16)(ck.f[rr] + bkc);
      vo[o] = (_Float16)(cv.f[rr] + bvc);
    }
  }
}

// ---------------- slot LayerNorm + Q projection (WMMA) ------------------------
__global__ __launch_bounds__(32) void kQProj(
    const float* __restrict__ slots, const float* g, const float* be,
    const _Float16* __restrict__ wq, const float* bq, _Float16* __restrict__ qo) {
  __shared__ _Float16 st[16][136];
  int lane = threadIdx.x;
  size_t row0 = (size_t)blockIdx.x * 16;
  int m = lane & 15, hh = lane >> 4;
  const float* r = slots + (row0 + m) * DIMD + hh * 64;
  float s = 0.f, ss = 0.f;
  #pragma unroll 8
  for (int i = 0; i < 64; ++i) { float t = r[i]; s += t; ss += t * t; }
  s  += __shfl_xor(s, 16, 32);
  ss += __shfl_xor(ss, 16, 32);
  float mean = s * (1.f / 128.f);
  float rstd = rsqrtf(ss * (1.f / 128.f) - mean * mean + 1e-5f);
  #pragma unroll 8
  for (int i = 0; i < 64; ++i) {
    int c = hh * 64 + i;
    st[m][c] = (_Float16)((r[i] - mean) * rstd * g[c] + be[c]);
  }
  __syncthreads();
  int kb8 = (lane < 16) ? 0 : 8;
  int kb16 = (lane < 16) ? 0 : 16;
  int n = lane & 15;
  AFrag af[4];
  #pragma unroll
  for (int kc = 0; kc < 4; ++kc)
    #pragma unroll
    for (int i = 0; i < 8; ++i) {
      af[kc].h[i]     = st[m][kc * 32 + kb16 + i];
      af[kc].h[8 + i] = st[m][kc * 32 + kb16 + 8 + i];
    }
  for (int nt = 0; nt < 8; ++nt) {
    CFrag c;
    #pragma unroll
    for (int i = 0; i < 8; ++i) c.f[i] = 0.f;
    #pragma unroll
    for (int kc = 0; kc < 4; ++kc) {
      BFrag bf;
      const _Float16* wp = wq + (size_t)(nt * 16 + n) * DIMD + kc * 32 + kb16;
      #pragma unroll
      for (int i = 0; i < 16; ++i) bf.h[i] = wp[i];
      c.v = wmma16(af[kc].v, bf.v, c.v);
    }
    float bqc = bq[nt * 16 + n];
    #pragma unroll
    for (int rr = 0; rr < 8; ++rr) {
      int mr = (lane < 16) ? rr : rr + 8;
      qo[(row0 + mr) * DIMD + nt * 16 + n] = (_Float16)(c.f[rr] + bqc);
    }
  }
}

// ---------------- dots = q k^T / TEMP -> cost(+noise) (WMMA) ------------------
// grid (256, 64): one wave per (batch, 16-column j-tile)
__global__ __launch_bounds__(32) void kDots(
    const _Float16* __restrict__ qbf, const _Float16* __restrict__ kbf,
    float* __restrict__ cost, int iter) {
  int lane = threadIdx.x;
  int jt = blockIdx.x, b = blockIdx.y;
  int m = lane & 15;
  int kb8 = (lane < 16) ? 0 : 8;
  int kb16 = (lane < 16) ? 0 : 16;
  int n = lane & 15;
  CFrag c;
  #pragma unroll
  for (int i = 0; i < 8; ++i) c.f[i] = 0.f;
  __builtin_prefetch(kbf + ((size_t)b * NSEQ + jt * 16 + 16 + n) * DIMD, 0, 0);
  #pragma unroll
  for (int kc = 0; kc < 4; ++kc) {
    AFrag a;
    if (m < KSLOT) {
      const _Float16* qp = qbf + ((size_t)b * KSLOT + m) * DIMD + kc * 32;
      #pragma unroll
      for (int i = 0; i < 8; ++i) {
        a.h[i]     = qp[kb16 + i];
        a.h[8 + i] = qp[kb16 + 8 + i];
      }
    } else {
      #pragma unroll
      for (int i = 0; i < 16; ++i) a.h[i] = (_Float16)0.f;
    }
    BFrag bf;
    const _Float16* kp = kbf + ((size_t)b * NSEQ + jt * 16 + n) * DIMD + kc * 32 + kb16;
    #pragma unroll
    for (int i = 0; i < 16; ++i) bf.h[i] = kp[i];
    c.v = wmma16(a.v, bf.v, c.v);
  }
  if (lane < 16) {
    int j = jt * 16 + n;
    float* cp = cost + ((size_t)b * NSEQ + j) * KSLOT;
    unsigned base = ((unsigned)(b * NSEQ + j) * 8u);
    #pragma unroll
    for (int rr = 0; rr < 8; ++rr) {
      float noise = 1e-4f * gaussf((base + rr) * 7u + (unsigned)iter + 1u);
      cp[rr] = -c.f[rr] * INV_TEMP + noise;
    }
  }
}

// ---------------- Sinkhorn: 30 iters held in registers per batch --------------
__device__ inline void lseComb(float& m, float& s, float m2, float s2) {
  float M = fmaxf(m, m2);
  s = s * __expf(m - M) + s2 * __expf(m2 - M);
  m = M;
}
__global__ __launch_bounds__(1024) void kSinkhorn(const float* __restrict__ cost,
                                                  _Float16* __restrict__ attnT) {
  __shared__ float redM[32][8];
  __shared__ float redS[32][8];
  __shared__ float lse[8];
  int b = blockIdx.x, tid = threadIdx.x;
  int lane = tid & 31, w = tid >> 5;
  const float* cb = cost + (size_t)b * NSEQ * KSLOT;
  int j0 = tid * 4;
  float lp[4][8];
  #pragma unroll
  for (int r = 0; r < 4; ++r)
    #pragma unroll
    for (int i = 0; i < 8; ++i)
      lp[r][i] = cb[(j0 + r) * 8 + i] * (-INV_REG);

  for (int it = 0; it < 30; ++it) {
    // logsumexp over K (axis 2): thread-local
    #pragma unroll
    for (int r = 0; r < 4; ++r) {
      float mx = lp[r][0];
      #pragma unroll
      for (int i = 1; i < 8; ++i) mx = fmaxf(mx, lp[r][i]);
      float sm = 0.f;
      #pragma unroll
      for (int i = 0; i < 8; ++i) sm += __expf(lp[r][i] - mx);
      float l = mx + __logf(sm);
      #pragma unroll
      for (int i = 0; i < 8; ++i) lp[r][i] -= l;
    }
    // logsumexp over N (axis 1): (max,sum) pair reduction
    float cm[8], cs[8];
    #pragma unroll
    for (int i = 0; i < 8; ++i) {
      float mx = lp[0][i];
      #pragma unroll
      for (int r = 1; r < 4; ++r) mx = fmaxf(mx, lp[r][i]);
      float sm = 0.f;
      #pragma unroll
      for (int r = 0; r < 4; ++r) sm += __expf(lp[r][i] - mx);
      cm[i] = mx; cs[i] = sm;
    }
    #pragma unroll
    for (int o = 16; o > 0; o >>= 1)
      #pragma unroll
      for (int i = 0; i < 8; ++i)
        lseComb(cm[i], cs[i], __shfl_xor(cm[i], o, 32), __shfl_xor(cs[i], o, 32));
    if (lane == 0)
      #pragma unroll
      for (int i = 0; i < 8; ++i) { redM[w][i] = cm[i]; redS[w][i] = cs[i]; }
    __syncthreads();
    if (tid < 8) {
      float M = redM[0][tid], S = redS[0][tid];
      for (int ww = 1; ww < 32; ++ww) lseComb(M, S, redM[ww][tid], redS[ww][tid]);
      lse[tid] = M + __logf(S);
    }
    __syncthreads();
    #pragma unroll
    for (int r = 0; r < 4; ++r)
      #pragma unroll
      for (int i = 0; i < 8; ++i) lp[r][i] -= lse[i];
    __syncthreads();
  }
  _Float16* at = attnT + (size_t)b * KSLOT * NSEQ;
  #pragma unroll
  for (int i = 0; i < 8; ++i)
    #pragma unroll
    for (int r = 0; r < 4; ++r)
      at[(size_t)i * NSEQ + j0 + r] = (_Float16)__expf(lp[r][i]);
}

// ---------------- updates = attn (K x N) @ v (N x D) --------------------------
// WMMA with double-buffered async global->LDS staging of v tiles.
// LDS tile: 32 rows x 136 f16 (row stride 272 B, 16B aligned for b128 async).
#define VROWH 136
#define VROWB (VROWH * 2)
#define VTILEB (32 * VROWB)
__global__ __launch_bounds__(32) void kUpdates(const _Float16* __restrict__ attnT,
                                               const _Float16* __restrict__ vbf,
                                               float* __restrict__ upd) {
  __shared__ __align__(16) _Float16 vt[2][32][VROWH];
  int lane = threadIdx.x;
  int b = blockIdx.x;
  int m = lane & 15;
  int kb8 = (lane < 16) ? 0 : 8;
  int kb16 = (lane < 16) ? 0 : 16;
  int n = lane & 15;
  CFrag acc[8];
  #pragma unroll
  for (int nt = 0; nt < 8; ++nt)
    #pragma unroll
    for (int i = 0; i < 8; ++i) acc[nt].f[i] = 0.f;
  const _Float16* vb = vbf + (size_t)b * NSEQ * DIMD;
  const _Float16* ab = attnT + (size_t)b * KSLOT * NSEQ;
  unsigned ldsBase = (unsigned)(size_t)(&vt[0][0][0]); // low 32 bits of flat addr = LDS offset

  // async-stage one 32x128 f16 tile (8 KB): 16 x b128 per wave, per-lane scatter
  auto issueTile = [&](int jc, int buf) {
    #pragma unroll
    for (int t = 0; t < 16; ++t) {
      int idx = t * 32 + lane;
      int row = idx >> 4, piece = idx & 15;
      unsigned lds = ldsBase + (unsigned)(buf * VTILEB + row * VROWB + piece * 16);
      unsigned long long ga =
          (unsigned long long)(const void*)(vb + ((size_t)(jc * 32 + row)) * DIMD + piece * 8);
      asm volatile("global_load_async_to_lds_b128 %0, %1, off"
                   :: "v"(lds), "v"(ga) : "memory");
    }
  };

  issueTile(0, 0);
  for (int jc = 0; jc < NSEQ / 32; ++jc) {
    int buf = jc & 1;
    if (jc + 1 < NSEQ / 32) {
      issueTile(jc + 1, buf ^ 1);
      asm volatile("s_wait_asynccnt 0x10" ::: "memory"); // tile jc resident; next in flight
    } else {
      asm volatile("s_wait_asynccnt 0x0" ::: "memory");
    }
    AFrag a;
    if (m < KSLOT) {
      const _Float16* ap = ab + (size_t)m * NSEQ + jc * 32;
      #pragma unroll
      for (int i = 0; i < 8; ++i) {
        a.h[i]     = ap[kb16 + i];
        a.h[8 + i] = ap[kb16 + 8 + i];
      }
    } else {
      #pragma unroll
      for (int i = 0; i < 16; ++i) a.h[i] = (_Float16)0.f;
    }
    #pragma unroll
    for (int nt = 0; nt < 8; ++nt) {
      BFrag bf;
      #pragma unroll
      for (int i = 0; i < 16; ++i) bf.h[i] = vt[buf][kb16 + i][nt * 16 + n];
      acc[nt].v = wmma16(a.v, bf.v, acc[nt].v);
    }
  }
  if (lane < 16) {
    #pragma unroll
    for (int nt = 0; nt < 8; ++nt)
      #pragma unroll
      for (int rr = 0; rr < 8; ++rr)
        upd[((size_t)b * KSLOT + rr) * DIMD + nt * 16 + n] = acc[nt].f[rr];
  }
}

// ---------------- GRU + FFN + MHA + LayerNorms on slots (per batch) -----------
__global__ __launch_bounds__(128) void kSlotUpdate(
    const float* __restrict__ slots_in, const float* __restrict__ updates,
    const float* W_ih, const float* b_ih, const float* W_hh, const float* b_hh,
    const float* preg, const float* preb,
    const float* W1, const float* b1, const float* W2, const float* b2,
    const float* Wqkv, const float* bqkv, const float* Wo, const float* bo,
    const float* lng, const float* lnb,
    float* __restrict__ slots_out, float* __restrict__ out) {
  __shared__ float s0[8][128], u[8][128], hcur[8][128], pre[8][128];
  __shared__ float g1[8][384], g2[8][384];
  __shared__ float hid[8][256];
  __shared__ float att[4][8][8];
  __shared__ float mv[8][2];
  int b = blockIdx.x, tid = threadIdx.x;
  const float* sb = slots_in + (size_t)b * 8 * 128;
  const float* ub = updates + (size_t)b * 8 * 128;
  for (int idx = tid; idx < 1024; idx += 128) {
    s0[idx >> 7][idx & 127] = sb[idx];
    u[idx >> 7][idx & 127] = ub[idx];
  }
  __syncthreads();
  // GRU gates
  for (int idx = tid; idx < 3072; idx += 128) {
    int i = idx / 384, c = idx % 384;
    const float* wi = W_ih + (size_t)c * 128;
    const float* wh = W_hh + (size_t)c * 128;
    float a1 = b_ih[c], a2 = b_hh[c];
    for (int d = 0; d < 128; ++d) { a1 += u[i][d] * wi[d]; a2 += s0[i][d] * wh[d]; }
    g1[i][c] = a1; g2[i][c] = a2;
  }
  __syncthreads();
  for (int idx = tid; idx < 1024; idx += 128) {
    int i = idx >> 7, c = idx & 127;
    float r = 1.f / (1.f + __expf(-(g1[i][c] + g2[i][c])));
    float z = 1.f / (1.f + __expf(-(g1[i][128 + c] + g2[i][128 + c])));
    float nn = tanhf(g1[i][256 + c] + r * g2[i][256 + c]);
    hcur[i][c] = (1.f - z) * nn + z * s0[i][c];
  }
  __syncthreads();
  // pre-FF layernorm
  if (tid < 8) {
    float s = 0.f, ss = 0.f;
    for (int d = 0; d < 128; ++d) { float t = hcur[tid][d]; s += t; ss += t * t; }
    float m = s / 128.f;
    mv[tid][0] = m; mv[tid][1] = rsqrtf(ss / 128.f - m * m + 1e-5f);
  }
  __syncthreads();
  for (int idx = tid; idx < 1024; idx += 128) {
    int i = idx >> 7, c = idx & 127;
    pre[i][c] = (hcur[i][c] - mv[i][0]) * mv[i][1] * preg[c] + preb[c];
  }
  __syncthreads();
  // FFN
  for (int idx = tid; idx < 2048; idx += 128) {
    int i = idx >> 8, c = idx & 255;
    const float* wptr = W1 + (size_t)c * 128;
    float a = b1[c];
    for (int d = 0; d < 128; ++d) a += pre[i][d] * wptr[d];
    hid[i][c] = fmaxf(a, 0.f);
  }
  __syncthreads();
  for (int idx = tid; idx < 1024; idx += 128) {
    int i = idx >> 7, c = idx & 127;
    const float* wptr = W2 + (size_t)c * 256;
    float a = b2[c];
    for (int d = 0; d < 256; ++d) a += hid[i][d] * wptr[d];
    hcur[i][c] += a;
  }
  __syncthreads();
  // MHA: qkv into g1
  for (int idx = tid; idx < 3072; idx += 128) {
    int i = idx / 384, c = idx % 384;
    const float* wptr = Wqkv + (size_t)c * 128;
    float a = bqkv[c];
    for (int d = 0; d < 128; ++d) a += hcur[i][d] * wptr[d];
    g1[i][c] = a;
  }
  __syncthreads();
  for (int idx = tid; idx < 256; idx += 128) {
    int hh = idx >> 6, rem = idx & 63, i = rem >> 3, j = rem & 7;
    float a = 0.f;
    for (int d = 0; d < 32; ++d) a += g1[i][hh * 32 + d] * g1[j][128 + hh * 32 + d];
    att[hh][i][j] = a * 0.17677669529663687f; // 1/sqrt(32)
  }
  __syncthreads();
  if (tid < 32) {
    int hh = tid >> 3, i = tid & 7;
    float mx = att[hh][i][0];
    for (int j = 1; j < 8; ++j) mx = fmaxf(mx, att[hh][i][j]);
    float s = 0.f;
    for (int j = 0; j < 8; ++j) { float e = __expf(att[hh][i][j] - mx); att[hh][i][j] = e; s += e; }
    float inv = 1.f / s;
    for (int j = 0; j < 8; ++j) att[hh][i][j] *= inv;
  }
  __syncthreads();
  for (int idx = tid; idx < 1024; idx += 128) {
    int i = idx >> 7, c = idx & 127, hh = c >> 5, d = c & 31;
    float a = 0.f;
    for (int j = 0; j < 8; ++j) a += att[hh][i][j] * g1[j][256 + hh * 32 + d];
    u[i][c] = a;
  }
  __syncthreads();
  for (int idx = tid; idx < 1024; idx += 128) {
    int i = idx >> 7, c = idx & 127;
    const float* wptr = Wo + (size_t)c * 128;
    float a = bo[c];
    for (int d = 0; d < 128; ++d) a += u[i][d] * wptr[d];
    pre[i][c] = hcur[i][c] + a;
  }
  __syncthreads();
  // final slot layernorm
  if (tid < 8) {
    float s = 0.f, ss = 0.f;
    for (int d = 0; d < 128; ++d) { float t = pre[tid][d]; s += t; ss += t * t; }
    float m = s / 128.f;
    mv[tid][0] = m; mv[tid][1] = rsqrtf(ss / 128.f - m * m + 1e-5f);
  }
  __syncthreads();
  for (int idx = tid; idx < 1024; idx += 128) {
    int i = idx >> 7, c = idx & 127;
    float vvv = (pre[i][c] - mv[i][0]) * mv[i][1] * lng[c] + lnb[c];
    slots_out[(size_t)b * 1024 + idx] = vvv;
    out[(size_t)b * 1024 + idx] = vvv;
  }
}

// ---------------- host launch --------------------------------------------------
extern "C" void kernel_launch(void* const* d_in, const int* in_sizes, int n_in,
                              void* d_out, int out_size, void* d_ws, size_t ws_size,
                              hipStream_t stream) {
  (void)in_sizes; (void)n_in; (void)out_size; (void)ws_size;
  const float* inputs   = (const float*)d_in[0];
  const float* slots_mu = (const float*)d_in[1];
  const float* log_sig  = (const float*)d_in[2];
  const float* Wq = (const float*)d_in[3];  const float* bq = (const float*)d_in[4];
  const float* Wk = (const float*)d_in[5];  const float* bk = (const float*)d_in[6];
  const float* Wv = (const float*)d_in[7];  const float* bv = (const float*)d_in[8];
  const float* W_ih = (const float*)d_in[9];  const float* b_ih = (const float*)d_in[10];
  const float* W_hh = (const float*)d_in[11]; const float* b_hh = (const float*)d_in[12];
  const float* W1 = (const float*)d_in[13]; const float* b1 = (const float*)d_in[14];
  const float* W2 = (const float*)d_in[15]; const float* b2 = (const float*)d_in[16];
  const float* Wqkv = (const float*)d_in[17]; const float* bqkv = (const float*)d_in[18];
  const float* Wo = (const float*)d_in[19]; const float* bo = (const float*)d_in[20];
  const float* ing = (const float*)d_in[21]; const float* inb = (const float*)d_in[22];
  const float* nsg = (const float*)d_in[23]; const float* nsb = (const float*)d_in[24];
  const float* pfg = (const float*)d_in[25]; const float* pfb = (const float*)d_in[26];
  const float* slg = (const float*)d_in[27]; const float* slb = (const float*)d_in[28];
  float* outp = (float*)d_out;

  // workspace carve-up
  _Float16* kbf   = (_Float16*)d_ws;                               // 64 MB
  _Float16* vbf   = kbf + (size_t)NB * NSEQ * DIMD;                // 64 MB
  float*    cost  = (float*)(vbf + (size_t)NB * NSEQ * DIMD);      // 8 MB
  _Float16* attnT = (_Float16*)(cost + (size_t)NB * NSEQ * KSLOT); // 4 MB
  float*    slots = (float*)(attnT + (size_t)NB * KSLOT * NSEQ);   // 256 KB
  float*    upd   = slots + (size_t)NB * KSLOT * DIMD;             // 256 KB
  _Float16* qbf   = (_Float16*)(upd + (size_t)NB * KSLOT * DIMD);  // 128 KB
  _Float16* wqh   = qbf + (size_t)NB * KSLOT * DIMD;
  _Float16* wkh   = wqh + DIMD * DIMD;
  _Float16* wvh   = wkh + DIMD * DIMD;

  kCvtW<<<(DIMD * DIMD + 255) / 256, 256, 0, stream>>>(Wq, Wk, Wv, wqh, wkh, wvh);
  kSlotInit<<<(NB * KSLOT * DIMD + 255) / 256, 256, 0, stream>>>(slots_mu, log_sig, slots);
  kProjKV<<<(NB * NSEQ) / 16, 32, 0, stream>>>(inputs, ing, inb, wkh, bk, wvh, bv, kbf, vbf);

  for (int it = 0; it < 3; ++it) {
    kQProj<<<(NB * KSLOT) / 16, 32, 0, stream>>>(slots, nsg, nsb, wqh, bq, qbf);
    kDots<<<dim3(NSEQ / 16, NB), 32, 0, stream>>>(qbf, kbf, cost, it);
    kSinkhorn<<<NB, 1024, 0, stream>>>(cost, attnT);
    kUpdates<<<NB, 32, 0, stream>>>(attnT, vbf, upd);
    kSlotUpdate<<<NB, 128, 0, stream>>>(slots, upd,
        W_ih, b_ih, W_hh, b_hh, pfg, pfb, W1, b1, W2, b2,
        Wqkv, bqkv, Wo, bo, slg, slb, slots, outp);
  }
}
